// MultiHeadAttention_64596308131860
// MI455X (gfx1250) — matrix operand order repaired and checked by reading, not compiled, not run on hardware
//
#include <hip/hip_runtime.h>
#include <hip/hip_bf16.h>

typedef __bf16 bf16;
typedef __attribute__((ext_vector_type(8)))  __bf16 v8bf;
typedef __attribute__((ext_vector_type(16))) __bf16 v16bf;
typedef __attribute__((ext_vector_type(8)))  float  v8f;
typedef __attribute__((ext_vector_type(4)))  int    v4i;

static constexpr int Bc  = 4;
static constexpr int Sc  = 2048;
static constexpr int Dc  = 512;
static constexpr int Hc  = 8;
static constexpr int DKc = 64;

// ---------------- CDNA5 async global->LDS helpers ----------------

#if defined(__has_builtin)
#if __has_builtin(__builtin_amdgcn_global_load_async_to_lds_b128)
#define HAVE_ASYNC_LDS 1
#endif
#if __has_builtin(__builtin_amdgcn_s_wait_asynccnt)
#define HAVE_WAIT_ASYNC 1
#endif
#endif
#ifndef HAVE_ASYNC_LDS
#define HAVE_ASYNC_LDS 0
#endif
#ifndef HAVE_WAIT_ASYNC
#define HAVE_WAIT_ASYNC 0
#endif

#if HAVE_ASYNC_LDS
typedef __attribute__((address_space(1))) v4i global_v4i;
typedef __attribute__((address_space(3))) v4i shared_v4i;
#endif

// 16-byte per-lane copy global -> LDS. Async path: no VGPR round trip, ASYNCcnt.
__device__ __forceinline__ void async_copy16(bf16* l, const bf16* g) {
#if HAVE_ASYNC_LDS
    __builtin_amdgcn_global_load_async_to_lds_b128((global_v4i*)g, (shared_v4i*)l,
                                                   0, 0);
#else
    *(v8bf*)l = *(const v8bf*)g;
#endif
}

template <int N>
__device__ __forceinline__ void wait_async() {
#if HAVE_WAIT_ASYNC
    __builtin_amdgcn_s_wait_asynccnt(N);
#else
    asm volatile("s_wait_asynccnt %0" ::"i"(N) : "memory");
#endif
}

// ---------------- WMMA helpers ----------------

__device__ __forceinline__ v8f wmma_bf16(v16bf a, v16bf b, v8f c) {
    return __builtin_amdgcn_wmma_f32_16x16x32_bf16(false, a, false, b, (short)0, c,
                                                   false, false);
}

// A fragment: 16x32 bf16, row-major source with leading dim `ld` (elements).
__device__ __forceinline__ v16bf load_frag_a(const bf16* base, int ld, int m0, int k0,
                                             int lane) {
    int half = lane >> 4;
    int row  = (lane & 15) + m0;
    const bf16* p = base + (size_t)row * ld + k0 + half * 8;
    v8bf lo = *(const v8bf*)(p);
    v8bf hi = *(const v8bf*)(p + 16);
    v16bf r;
#pragma unroll
    for (int j = 0; j < 8; ++j) { r[j] = lo[j]; r[j + 8] = hi[j]; }
    return r;
}

// B fragment: 32x16 bf16 sourced from TRANSPOSED storage Bt[n][k], ld in elements.
__device__ __forceinline__ v16bf load_frag_b(const bf16* baseT, int ld, int n0, int k0,
                                             int lane) {
    int half = lane >> 4;
    int col  = (lane & 15) + n0;
    const bf16* p = baseT + (size_t)col * ld + k0 + half * 8;
    v8bf lo = *(const v8bf*)(p);
    v8bf hi = *(const v8bf*)(p + 16);
    v16bf r;
#pragma unroll
    for (int j = 0; j < 8; ++j) { r[j] = lo[j]; r[j + 8] = hi[j]; }
    return r;
}

// ---------------- conversion kernels ----------------

__global__ void convert_f32_bf16(const float* __restrict__ in, bf16* __restrict__ out,
                                 int n) {
    int i = blockIdx.x * blockDim.x + threadIdx.x;
    if (i < n) out[i] = (bf16)in[i];
}

// in: [z][rows][cols] f32 -> out: [z][cols][rows] bf16
__global__ void transpose_f32_bf16(const float* __restrict__ in, bf16* __restrict__ out,
                                   int rows, int cols) {
    size_t zoff = (size_t)blockIdx.z * rows * cols;
    int c = blockIdx.x * 16 + threadIdx.x;
    int r = blockIdx.y * 16 + threadIdx.y;
    if (r < rows && c < cols)
        out[zoff + (size_t)c * rows + r] = (bf16)in[zoff + (size_t)r * cols + c];
}

// ---------------- QKV projection: Q/K[b][h][s][dk], V transposed [b][h][dk][s] ----

__global__ __launch_bounds__(256) void qkv_gemm(const bf16* __restrict__ Xb,
                                                const bf16* __restrict__ Wqt,
                                                const bf16* __restrict__ Wkt,
                                                const bf16* __restrict__ Wvt,
                                                bf16* __restrict__ Q,
                                                bf16* __restrict__ K,
                                                bf16* __restrict__ Vt) {
    __shared__ __align__(16) bf16 Xs[2][64 * 40];   // 64 rows x 32 k (pad 40), x2 buf
    __shared__ __align__(32) bf16 Ws[2][64 * 48];   // 64 n x 32 k (pad 48), transposed

    int mb    = blockIdx.x * 64;
    int h     = blockIdx.y / 3;
    int which = blockIdx.y % 3;
    int b     = blockIdx.z;

    const bf16* WT = (which == 0 ? Wqt : which == 1 ? Wkt : Wvt) + (size_t)h * DKc * Dc;
    const bf16* Xp = Xb + ((size_t)b * Sc + mb) * Dc;

    int tid = threadIdx.x, lane = tid & 31, wave = tid >> 5;
    int row = tid >> 2, seg = tid & 3;
    const bf16* xsrc = Xp + (size_t)row * Dc + seg * 8;   // + kb per step
    const bf16* wsrc = WT + (size_t)row * Dc + seg * 8;
    int xofs = row * 40 + seg * 8;
    int wofs = row * 48 + seg * 8;

    // preload k-chunk 0 into buffer 0
    async_copy16(&Xs[0][xofs], xsrc);
    async_copy16(&Ws[0][wofs], wsrc);

    v8f acc[2] = {};
    int cur = 0;
    for (int kb = 0; kb < Dc; kb += 32, cur ^= 1) {
        if (kb + 32 < Dc) {            // prefetch next chunk into other buffer
            async_copy16(&Xs[cur ^ 1][xofs], xsrc + kb + 32);
            async_copy16(&Ws[cur ^ 1][wofs], wsrc + kb + 32);
            wait_async<2>();           // drain current chunk's (in-order) loads
        } else {
            wait_async<0>();
        }
        __syncthreads();
#pragma unroll
        for (int i = 0; i < 2; ++i) {
            int tile = wave * 2 + i, mt = tile >> 2, nt = tile & 3;
            v16bf a  = load_frag_a(&Xs[cur][0], 40, mt * 16, 0, lane);
            v16bf bb = load_frag_b(&Ws[cur][0], 48, nt * 16, 0, lane);
            acc[i]   = wmma_bf16(a, bb, acc[i]);
        }
        __syncthreads();               // protect buffer reuse next iteration
    }

    int half = lane >> 4, col = lane & 15;
#pragma unroll
    for (int i = 0; i < 2; ++i) {
        int tile = wave * 2 + i, mt = tile >> 2, nt = tile & 3;
        int rowbase = mb + mt * 16 + half * 8;
        int n       = nt * 16 + col;
        if (which == 2) {
            v8bf pk;
#pragma unroll
            for (int r = 0; r < 8; ++r) pk[r] = (bf16)acc[i][r];
            *(v8bf*)&Vt[(((size_t)b * Hc + h) * DKc + n) * Sc + rowbase] = pk;
        } else {
            bf16* O = (which == 0 ? Q : K) + ((size_t)b * Hc + h) * Sc * DKc;
#pragma unroll
            for (int r = 0; r < 8; ++r)
                O[(size_t)(rowbase + r) * DKc + n] = (bf16)acc[i][r];
        }
    }
}

// ---------------- fused causal flash attention -------------------------------
// Block: 128 query rows, 8 waves x 16 rows. Oc[b][s][dk*H + h] (reference concat)

__global__ __launch_bounds__(256) void flash_attn(const bf16* __restrict__ Q,
                                                  const bf16* __restrict__ Km,
                                                  const bf16* __restrict__ Vt,
                                                  bf16* __restrict__ Oc) {
    __shared__ __align__(32) bf16 Ks[2][64 * 80];    // keys x dk, double buffered
    __shared__ __align__(32) bf16 Vs[2][64 * 80];    // dk x keys (Bt form for P*V)
    __shared__ __align__(16) bf16 Pst[8 * 16 * 72];  // per-wave P staging (A-layout)

    int qb = blockIdx.x * 128;
    int h  = blockIdx.y;
    int b  = blockIdx.z;
    int tid = threadIdx.x, lane = tid & 31, wave = tid >> 5;
    int half = lane >> 4, col = lane & 15;
    size_t bh = (size_t)b * Hc + h;

    int q0 = qb + wave * 16;
    const bf16* Qp = Q + (bh * Sc + q0) * DKc;
    v16bf qa0 = load_frag_a(Qp, DKc, 0, 0, lane);
    v16bf qa1 = load_frag_a(Qp, DKc, 0, 32, lane);

    v8f oacc[4] = {};
    float mrow[8], lrow[8];
#pragma unroll
    for (int r = 0; r < 8; ++r) { mrow[r] = -1e30f; lrow[r] = 0.0f; }

    bf16* Pw = &Pst[wave * 16 * 72];
    int nkb = 2 * blockIdx.x + 2;   // causal: kv blocks with kv0 <= qb+127

    // per-thread source/dest for the cooperative 64x64 tile copies (2x16B each)
    auto issueKV = [&](int buf, int kv0) {
#pragma unroll
        for (int it = 0; it < 2; ++it) {
            int idx = tid + it * 256;
            int r = idx >> 3, sg = idx & 7;
            async_copy16(&Ks[buf][r * 80 + sg * 8],
                         &Km[(bh * Sc + kv0 + r) * DKc + sg * 8]);
            async_copy16(&Vs[buf][r * 80 + sg * 8],
                         &Vt[(bh * DKc + r) * Sc + kv0 + sg * 8]);
        }
    };

    issueKV(0, 0);
    int cur = 0;
    for (int kb = 0; kb < nkb; ++kb, cur ^= 1) {
        int kv0 = kb * 64;
        if (kb + 1 < nkb) {
            issueKV(cur ^ 1, kv0 + 64);  // prefetch next KV tile during compute
            wait_async<4>();             // drain current tile's 4 in-order loads
        } else {
            wait_async<0>();
        }
        __syncthreads();

        // S = Q * K^T  (16 q-rows x 64 keys per wave)
        v8f sacc[4] = {};
#pragma unroll
        for (int nt = 0; nt < 4; ++nt) {
            v16bf b0 = load_frag_b(&Ks[cur][0], 80, nt * 16, 0, lane);
            sacc[nt] = wmma_bf16(qa0, b0, sacc[nt]);
            v16bf b1 = load_frag_b(&Ks[cur][0], 80, nt * 16, 32, lane);
            sacc[nt] = wmma_bf16(qa1, b1, sacc[nt]);
        }

        bool needMask = (kv0 + 63) > q0;
#pragma unroll
        for (int r = 0; r < 8; ++r) {
            int rowg = q0 + half * 8 + r;
            float s[4];
            float mx = -1e30f;
#pragma unroll
            for (int nt = 0; nt < 4; ++nt) {
                float v = sacc[nt][r] * 0.125f;                 // 1/sqrt(64)
                if (needMask && (kv0 + nt * 16 + col) > rowg) v = -1e30f;
                s[nt] = v;
                mx = fmaxf(mx, v);
            }
#pragma unroll
            for (int d = 1; d < 16; d <<= 1) mx = fmaxf(mx, __shfl_xor(mx, d, 32));
            float mnew  = fmaxf(mrow[r], mx);
            float alpha = __expf(mrow[r] - mnew);
            float rs = 0.0f;
#pragma unroll
            for (int nt = 0; nt < 4; ++nt) {
                float p = __expf(s[nt] - mnew);
                rs += p;
                Pw[(half * 8 + r) * 72 + nt * 16 + col] = (bf16)p;
            }
#pragma unroll
            for (int d = 1; d < 16; d <<= 1) rs += __shfl_xor(rs, d, 32);
            lrow[r] = lrow[r] * alpha + rs;
            mrow[r] = mnew;
#pragma unroll
            for (int nt = 0; nt < 4; ++nt) oacc[nt][r] *= alpha;
        }

        // O += P * V   (per-wave LDS staging; same-wave LDS ops are in-order)
        v16bf pa0 = load_frag_a(Pw, 72, 0, 0, lane);
        v16bf pa1 = load_frag_a(Pw, 72, 0, 32, lane);
#pragma unroll
        for (int nt = 0; nt < 4; ++nt) {
            v16bf v0 = load_frag_b(&Vs[cur][0], 80, nt * 16, 0, lane);
            oacc[nt] = wmma_bf16(pa0, v0, oacc[nt]);
            v16bf v1 = load_frag_b(&Vs[cur][0], 80, nt * 16, 32, lane);
            oacc[nt] = wmma_bf16(pa1, v1, oacc[nt]);
        }
        __syncthreads();   // all waves done with buffer `cur` before its re-fill
    }

    // epilogue: divide by softmax denom, write concat order c = dk*H + h
#pragma unroll
    for (int r = 0; r < 8; ++r) {
        float inv = 1.0f / lrow[r];
        int rowg  = q0 + half * 8 + r;
#pragma unroll
        for (int nt = 0; nt < 4; ++nt) {
            int dk = nt * 16 + col;
            Oc[((size_t)b * Sc + rowg) * (Hc * DKc) + (size_t)dk * Hc + h] =
                (bf16)(oacc[nt][r] * inv);
        }
    }
}

// ---------------- output projection: [B*S,512] x [512,512] -> f32 ------------

__global__ __launch_bounds__(256) void out_proj(const bf16* __restrict__ Oc,
                                                const bf16* __restrict__ WoT,
                                                float* __restrict__ out) {
    __shared__ __align__(16) bf16 As[2][64 * 40];
    __shared__ __align__(32) bf16 Bs[2][64 * 48];

    int mb = blockIdx.x * 64, nb = blockIdx.y * 64;
    int tid = threadIdx.x, lane = tid & 31, wave = tid >> 5;
    int row = tid >> 2, seg = tid & 3;
    const bf16* asrc = Oc + (size_t)(mb + row) * 512 + seg * 8;
    const bf16* bsrc = WoT + (size_t)(nb + row) * 512 + seg * 8;
    int aofs = row * 40 + seg * 8;
    int bofs = row * 48 + seg * 8;

    async_copy16(&As[0][aofs], asrc);
    async_copy16(&Bs[0][bofs], bsrc);

    v8f acc[2] = {};
    int cur = 0;
    for (int kb = 0; kb < 512; kb += 32, cur ^= 1) {
        if (kb + 32 < 512) {
            async_copy16(&As[cur ^ 1][aofs], asrc + kb + 32);
            async_copy16(&Bs[cur ^ 1][bofs], bsrc + kb + 32);
            wait_async<2>();
        } else {
            wait_async<0>();
        }
        __syncthreads();
#pragma unroll
        for (int i = 0; i < 2; ++i) {
            int tile = wave * 2 + i, mt = tile >> 2, nt = tile & 3;
            v16bf a  = load_frag_a(&As[cur][0], 40, mt * 16, 0, lane);
            v16bf bb = load_frag_b(&Bs[cur][0], 48, nt * 16, 0, lane);
            acc[i]   = wmma_bf16(a, bb, acc[i]);
        }
        __syncthreads();
    }

    int half = lane >> 4, col = lane & 15;
#pragma unroll
    for (int i = 0; i < 2; ++i) {
        int tile = wave * 2 + i, mt = tile >> 2, nt = tile & 3;
        int rowg = mb + mt * 16 + half * 8;
        int n    = nb + nt * 16 + col;
#pragma unroll
        for (int r = 0; r < 8; ++r)
            out[(size_t)(rowg + r) * 512 + n] = acc[i][r];
    }
}

// ---------------- launch ------------------------------------------------------

extern "C" void kernel_launch(void* const* d_in, const int* in_sizes, int n_in,
                              void* d_out, int out_size, void* d_ws, size_t ws_size,
                              hipStream_t stream) {
    (void)in_sizes; (void)n_in; (void)out_size; (void)ws_size;

    const float* X  = (const float*)d_in[0];
    const float* Wq = (const float*)d_in[1];
    const float* Wk = (const float*)d_in[2];
    const float* Wv = (const float*)d_in[3];
    const float* Wo = (const float*)d_in[4];
    float* out = (float*)d_out;

    size_t off = 0;
    auto carve = [&](size_t nelem) {
        bf16* p = (bf16*)((char*)d_ws + off);
        off = (off + nelem * sizeof(bf16) + 255) & ~(size_t)255;
        return p;
    };
    bf16* Xb  = carve((size_t)Bc * Sc * Dc);
    bf16* Wqt = carve((size_t)Hc * Dc * DKc);
    bf16* Wkt = carve((size_t)Hc * Dc * DKc);
    bf16* Wvt = carve((size_t)Hc * Dc * DKc);
    bf16* Wot = carve((size_t)Dc * Hc * DKc);
    bf16* Q   = carve((size_t)Bc * Hc * Sc * DKc);
    bf16* K   = carve((size_t)Bc * Hc * Sc * DKc);
    bf16* Vt  = carve((size_t)Bc * Hc * Sc * DKc);
    bf16* Oc  = carve((size_t)Bc * Sc * Hc * DKc);

    // 1) precision conversion + weight transposes
    int nx = Bc * Sc * Dc;
    convert_f32_bf16<<<(nx + 255) / 256, 256, 0, stream>>>(X, Xb, nx);
    transpose_f32_bf16<<<dim3(DKc / 16, Dc / 16, Hc), dim3(16, 16), 0, stream>>>(Wq, Wqt, Dc, DKc);
    transpose_f32_bf16<<<dim3(DKc / 16, Dc / 16, Hc), dim3(16, 16), 0, stream>>>(Wk, Wkt, Dc, DKc);
    transpose_f32_bf16<<<dim3(DKc / 16, Dc / 16, Hc), dim3(16, 16), 0, stream>>>(Wv, Wvt, Dc, DKc);
    transpose_f32_bf16<<<dim3(Dc / 16, (Hc * DKc) / 16, 1), dim3(16, 16), 0, stream>>>(Wo, Wot, Hc * DKc, Dc);

    // 2) Q/K/V projections (WMMA, async double-buffered LDS)
    qkv_gemm<<<dim3(Sc / 64, 3 * Hc, Bc), 256, 0, stream>>>(Xb, Wqt, Wkt, Wvt, Q, K, Vt);

    // 3) fused causal flash attention (WMMA + online softmax + async KV pipeline)
    flash_attn<<<dim3(Sc / 128, Hc, Bc), 256, 0, stream>>>(Q, K, Vt, Oc);

    // 4) output projection (WMMA), f32 result
    out_proj<<<dim3((Bc * Sc) / 64, Dc / 64, 1), 256, 0, stream>>>(Oc, Wot, out);
}
